// SpearmanCorrelationLoss_67757404062132
// MI455X (gfx1250) — 8-bit matrix operand order repaired and checked by reading, compiled, not run on hardware
//
#include <hip/hip_runtime.h>
#include <stdint.h>

typedef int v8i __attribute__((ext_vector_type(8)));

#define N_ROWS   16384
#define N_COLS   512
#define NTHREADS 1024
#define EPS_D    1e-6

__device__ __forceinline__ uint32_t float_to_ordered(float f) {
    uint32_t m = __float_as_uint(f);
    uint32_t s = (uint32_t)(-(int32_t)(m >> 31));
    return m ^ (s | 0x80000000u);
}

__global__ __launch_bounds__(NTHREADS)
void spearman_col_kernel(const float* __restrict__ pred,
                         const float* __restrict__ target,
                         double* __restrict__ corr_out) {
    __shared__ __align__(16) unsigned long long keys[N_ROWS];
    __shared__ __align__(16) unsigned short     rankp[N_ROWS];
    __shared__ __align__(16) unsigned short     rankt[N_ROWS];
    __shared__ int red[96];

    const int tid = threadIdx.x;
    const int col = blockIdx.x;

    for (int pass = 0; pass < 2; ++pass) {
        const float* src = pass ? target : pred;

        for (int s = 0; s < N_ROWS / NTHREADS; ++s) {
            int i = s * NTHREADS + tid;
            float v = src[(size_t)i * N_COLS + col];
            keys[i] = ((unsigned long long)float_to_ordered(v) << 32) |
                      (unsigned long long)(unsigned)i;
        }
        __syncthreads();

        for (unsigned k = 2; k <= N_ROWS; k <<= 1) {
            for (unsigned j = k >> 1; j >= 1; j >>= 1) {
                for (unsigned p = tid; p < N_ROWS / 2; p += NTHREADS) {
                    unsigned i   = ((p & ~(j - 1)) << 1) | (p & (j - 1));
                    unsigned ixj = i | j;
                    bool up = ((i & k) == 0);
                    unsigned long long a = keys[i];
                    unsigned long long b = keys[ixj];
                    bool sw = up ? (a > b) : (a < b);
                    if (sw) { keys[i] = b; keys[ixj] = a; }
                }
                __syncthreads();
            }
        }

        unsigned short* rk = pass ? rankt : rankp;
        for (int p = tid; p < N_ROWS; p += NTHREADS) {
            unsigned r = (unsigned)(keys[p] & 0xFFFFFFFFull);
            rk[r] = (unsigned short)(p + 1);
        }
        __syncthreads();
    }

    if (tid < 32) {
        const int lane = tid;
        const int half = lane >> 4;
        const int mn   = lane & 15;
        const unsigned long long* rp64 = (const unsigned long long*)rankp;
        const unsigned long long* rt64 = (const unsigned long long*)rankt;

        v8i chh = {}; v8i chl = {}; v8i clh = {}; v8i cll = {};

        for (int iter = 0; iter < 16; ++iter) {
            const int base = iter * 1024;
            v8i a_hi, a_lo, b_hi, b_lo;
#pragma unroll
            for (int v = 0; v < 8; ++v) {
                int ka = ((v >> 1) << 4) | ((v & 1) << 2) | (half << 3);
                {
                    int idx = base + mn * 64 + ka;
                    unsigned long long q = rp64[idx >> 2];
                    uint32_t hi = 0, lo = 0;
#pragma unroll
                    for (int b = 0; b < 4; ++b) {
                        uint32_t r = (uint32_t)((q >> (16 * b)) & 0xFFFFu);
                        lo |= (r & 127u) << (8 * b);
                        hi |= (r >> 7)   << (8 * b);
                    }
                    a_hi[v] = (int)hi; a_lo[v] = (int)lo;
                }
                int kb = ((v >> 1) << 4) | ((v & 1) << 2) | (half << 3);
                {
                    int idx = base + mn * 64 + kb;
                    unsigned long long q = rt64[idx >> 2];
                    uint32_t hi = 0, lo = 0;
#pragma unroll
                    for (int b = 0; b < 4; ++b) {
                        uint32_t r = (uint32_t)((q >> (16 * b)) & 0xFFFFu);
                        lo |= (r & 127u) << (8 * b);
                        hi |= (r >> 7)   << (8 * b);
                    }
                    b_hi[v] = (int)hi; b_lo[v] = (int)lo;
                }
            }
            chh = __builtin_amdgcn_wmma_i32_16x16x64_iu8(false, a_hi, false, b_hi, chh, false, false);
            chl = __builtin_amdgcn_wmma_i32_16x16x64_iu8(false, a_hi, false, b_lo, chl, false, false);
            clh = __builtin_amdgcn_wmma_i32_16x16x64_iu8(false, a_lo, false, b_hi, clh, false, false);
            cll = __builtin_amdgcn_wmma_i32_16x16x64_iu8(false, a_lo, false, b_lo, cll, false, false);
        }

        int hh = 0, hl = 0, ll = 0;
        if (lane < 8) {
            hh = chh[lane]; hl = chl[lane] + clh[lane]; ll = cll[lane];
        } else if (lane >= 24) {
            int g = lane - 24;
            hh = chh[g]; hl = chl[g] + clh[g]; ll = cll[g];
        }
        red[lane]      = hh;
        red[32 + lane] = hl;
        red[64 + lane] = ll;
    }
    __syncthreads();

    if (tid == 0) {
        long long HH = 0, HL = 0, LL = 0;
        for (int l = 0; l < 32; ++l) {
            HH += red[l]; HL += red[32 + l]; LL += red[64 + l];
        }
        unsigned long long S = ((unsigned long long)HH << 14) +
                               ((unsigned long long)HL << 7) +
                               (unsigned long long)LL;
        const double mu  = (double)(N_ROWS + 1) * 0.5;
        const double var = ((double)N_ROWS * (double)N_ROWS - 1.0) / 12.0;
        double cov  = (double)S / (double)N_ROWS - mu * mu;
        double corr = cov / ((var + EPS_D) + EPS_D);
        corr_out[col] = corr;
    }
}

__global__ __launch_bounds__(N_COLS)
void spearman_reduce_kernel(const double* __restrict__ corr,
                            float* __restrict__ out) {
    __shared__ double sdata[N_COLS];
    int t = threadIdx.x;
    sdata[t] = corr[t];
    __syncthreads();
    for (int off = N_COLS / 2; off > 0; off >>= 1) {
        if (t < off) sdata[t] += sdata[t + off];
        __syncthreads();
    }
    if (t == 0) out[0] = (float)(-sdata[0] / (double)N_COLS);
}

extern "C" void kernel_launch(void* const* d_in, const int* in_sizes, int n_in,
                              void* d_out, int out_size, void* d_ws, size_t ws_size,
                              hipStream_t stream) {
    const float* pred   = (const float*)d_in[0];
    const float* target = (const float*)d_in[1];
    double* corr = (double*)d_ws;

    spearman_col_kernel<<<N_COLS, NTHREADS, 0, stream>>>(pred, target, corr);
    spearman_reduce_kernel<<<1, N_COLS, 0, stream>>>(corr, (float*)d_out);
}
